// GAT_DSSE_BiLevel_37211596652682
// MI455X (gfx1250) — compile-verified, matrix operand order checked by reading, not run on hardware
//
#include <hip/hip_runtime.h>
#include <hip/hip_bf16.h>

// ---------------------------------------------------------------------------
// GATv2 (3 layers, H=4, C=64, edge_dim=16) + dense head, MI455X / gfx1250.
// Dense matmuls: v_wmma_f32_16x16x32_f16 with pre-converted f16 operands,
// branch-free b128 operand loads. Edge phase: wave-per-edge float4 gathers,
// LDS-resident edge weights, f32 atomics for segment softmax / aggregation.
// ---------------------------------------------------------------------------

typedef __attribute__((ext_vector_type(16))) _Float16 v16h;
typedef __attribute__((ext_vector_type(8)))  _Float16 v8h;
typedef __attribute__((ext_vector_type(8)))  float    v8f;

#define N_NODES   30000
#define N_EDGES   480000
#define E_TOT     (N_EDGES + N_NODES)   // 510000
#define HC        256                   // HEADS * DIM_HIDDEN
#define EDGE_DIM  16
#define NEG_SLOPE 0.2f
#define ACT_SLOPE 0.01f

// ---------------------------------------------------------------------------
// Generic fill / f32->f16 convert / weight convert+transpose (zero-padded)
// ---------------------------------------------------------------------------
__global__ void k_fill_f32(float* __restrict__ p, float v, int n) {
  int t = blockIdx.x * 256 + threadIdx.x;
  if (t < n) p[t] = v;
}

__global__ void k_x_to_f16(const float* __restrict__ X, _Float16* __restrict__ Xh,
                           int n) {
  int t = blockIdx.x * 256 + threadIdx.x;
  if (t < n) Xh[t] = (_Float16)X[t];
}

// Wt[n, k] = W[k, n] (f16), n in [0, Npad), zero-padded beyond N.
__global__ void k_w_to_f16t(const float* __restrict__ W, _Float16* __restrict__ Wt,
                            int K, int N, int Npad) {
  int t = blockIdx.x * 256 + threadIdx.x;
  if (t >= Npad * K) return;
  int n = t / K, k = t - n * K;
  Wt[t] = (n < N) ? (_Float16)W[k * N + n] : (_Float16)0.0f;
}

// ---------------------------------------------------------------------------
// WMMA GEMM: Y[M,Nout] = act( Xh[M,K] @ Wt[Npad,K]^T + bias )
// M % 16 == 0, K % 32 == 0 (compile-time), Npad % 64 == 0.
// One 16x16 tile per wave; 4 waves/block cover 4 adjacent N-tiles.
// All operand loads are unguarded 16B vector loads.
// ---------------------------------------------------------------------------
template<int K>
__global__ __launch_bounds__(128)
void k_wmma_gemm(const _Float16* __restrict__ Xh, const _Float16* __restrict__ Wt,
                 const float* __restrict__ bias, float* __restrict__ Y,
                 int Nout, int act_mode) {
  const int lane = threadIdx.x & 31;
  const int wid  = threadIdx.x >> 5;
  const int tm   = blockIdx.x * 16;
  const int tn   = (blockIdx.y * 4 + wid) * 16;
  const int r    = lane & 15;
  const int row  = tm + r;              // A row for this lane
  const int col  = tn + r;              // B/D column for this lane
  const int kg   = (lane >> 4) * 8;     // K sub-group: lanes 0-15 -> 0, 16-31 -> 8

  const _Float16* __restrict__ arow = Xh + (size_t)row * K;
  const _Float16* __restrict__ bcol = Wt + (size_t)col * K;

  v8f acc = {};
#pragma unroll
  for (int k0 = 0; k0 < K; k0 += 32) {
    v8h a0 = *(const v8h*)(arow + k0 + kg);        // halves 0..7 : K=kg+0..7
    v8h a1 = *(const v8h*)(arow + k0 + 16 + kg);   // halves 8..15: K=16+kg+0..7
    v8h b0 = *(const v8h*)(bcol + k0 + kg);
    v8h b1 = *(const v8h*)(bcol + k0 + 16 + kg);
    v16h a = __builtin_shufflevector(a0, a1, 0, 1, 2, 3, 4, 5, 6, 7,
                                     8, 9, 10, 11, 12, 13, 14, 15);
    v16h b = __builtin_shufflevector(b0, b1, 0, 1, 2, 3, 4, 5, 6, 7,
                                     8, 9, 10, 11, 12, 13, 14, 15);
    acc = __builtin_amdgcn_wmma_f32_16x16x32_f16(
        /*neg_a=*/false, a, /*neg_b=*/false, b,
        /*c_mod=*/(short)0, acc, /*reuse_a=*/false, /*reuse_b=*/false);
  }

  if (col >= Nout) return;
  const float bv = bias[col];
  const int mh = (lane >> 4) * 8;       // D: lanes 0-15 -> M=i, 16-31 -> M=8+i
#pragma unroll
  for (int i = 0; i < 8; ++i) {
    int m = tm + mh + i;
    float v = acc[i] + bv;
    if (act_mode == 1) v = (v >= 0.0f) ? v : ACT_SLOPE * v;
    Y[(size_t)m * Nout + col] = v;
  }
}

// ---------------------------------------------------------------------------
// Edge prep: copy edge_attr into ea_all[0:E), scatter sums into ea_all tail
// (self-loop mean attrs), count incoming edges, fill src/dst arrays.
// ---------------------------------------------------------------------------
__global__ void k_prep_edges(const int* __restrict__ ei,
                             const float* __restrict__ eattr,
                             float* __restrict__ ea_all,
                             int* __restrict__ srcA, int* __restrict__ dstA,
                             float* __restrict__ cnt) {
  int t = blockIdx.x * 256 + threadIdx.x;
  if (t >= N_EDGES * EDGE_DIM) return;
  int e = t >> 4, k = t & 15;
  int d = ei[N_EDGES + e];
  float v = eattr[t];
  ea_all[t] = v;
  atomicAdd(&ea_all[(N_EDGES + d) * EDGE_DIM + k], v);
  if (k == 0) {
    srcA[e] = ei[e];
    dstA[e] = d;
    atomicAdd(&cnt[d], 1.0f);
  }
}

__global__ void k_prep_loops(float* __restrict__ ea_all,
                             int* __restrict__ srcA, int* __restrict__ dstA,
                             const float* __restrict__ cnt) {
  int t = blockIdx.x * 256 + threadIdx.x;
  if (t >= N_NODES * EDGE_DIM) return;
  int n = t >> 4, k = t & 15;
  ea_all[(N_EDGES + n) * EDGE_DIM + k] /= fmaxf(cnt[n], 1.0f);
  if (k == 0) { srcA[N_EDGES + n] = n; dstA[N_EDGES + n] = n; }
}

// ---------------------------------------------------------------------------
// Pass A: per-edge logits (el recomputed from LDS We) + atomic segment max.
// One wave per edge; lane handles 8 contiguous channels (two float4 rows).
// ---------------------------------------------------------------------------
__device__ inline void atomicMaxF(float* addr, float val) {
  unsigned int* a = (unsigned int*)addr;
  unsigned int old = *a;
  while (__uint_as_float(old) < val) {
    unsigned int assumed = old;
    old = atomicCAS(a, assumed, __float_as_uint(val));
    if (old == assumed) break;
  }
}

__global__ __launch_bounds__(256)
void k_edge_logits(const float* __restrict__ xl, const float* __restrict__ xr,
                   const float* __restrict__ ea_all,
                   const int* __restrict__ srcA, const int* __restrict__ dstA,
                   const float* __restrict__ We, const float* __restrict__ att,
                   float* __restrict__ logits, float* __restrict__ mx) {
  __shared__ float sWe[EDGE_DIM * HC];   // 16 KB
  __shared__ float sAtt[HC];             // 1 KB
  for (int i = threadIdx.x; i < EDGE_DIM * HC; i += 256) sWe[i] = We[i];
  if (threadIdx.x < HC) sAtt[threadIdx.x] = att[threadIdx.x];
  __syncthreads();

  const int e    = (blockIdx.x * 256 + threadIdx.x) >> 5;   // global wave id
  const int lane = threadIdx.x & 31;
  if (e >= E_TOT) return;
  const int s = srcA[e], d = dstA[e];

  // 16 edge attrs as 4x float4
  const float4* eap = (const float4*)(ea_all + (size_t)e * EDGE_DIM);
  float4 ea0 = eap[0], ea1 = eap[1], ea2 = eap[2], ea3 = eap[3];
  float eav[EDGE_DIM] = {ea0.x, ea0.y, ea0.z, ea0.w, ea1.x, ea1.y, ea1.z, ea1.w,
                         ea2.x, ea2.y, ea2.z, ea2.w, ea3.x, ea3.y, ea3.z, ea3.w};

  // this lane's 8 channels of xl[src], xr[dst]
  const float4* xlp = (const float4*)(xl + (size_t)s * HC) + lane * 2;
  const float4* xrp = (const float4*)(xr + (size_t)d * HC) + lane * 2;
  float4 l0 = xlp[0], l1 = xlp[1];
  float4 r0 = xrp[0], r1 = xrp[1];
  float lv[8] = {l0.x, l0.y, l0.z, l0.w, l1.x, l1.y, l1.z, l1.w};
  float rv[8] = {r0.x, r0.y, r0.z, r0.w, r1.x, r1.y, r1.z, r1.w};

  float part = 0.0f;
  const int cbase = lane * 8;
#pragma unroll
  for (int j = 0; j < 8; ++j) {
    int c = cbase + j;
    float el = 0.0f;
#pragma unroll
    for (int k = 0; k < EDGE_DIM; ++k) el += eav[k] * sWe[k * HC + c];
    float m = lv[j] + rv[j] + el;
    m = (m >= 0.0f) ? m : NEG_SLOPE * m;
    part += m * sAtt[c];
  }
  // reduce within each group of 8 lanes (one head per group)
  part += __shfl_xor(part, 1, 32);
  part += __shfl_xor(part, 2, 32);
  part += __shfl_xor(part, 4, 32);
  if ((lane & 7) == 0) {
    int h = lane >> 3;
    logits[e * 4 + h] = part;
    atomicMaxF(&mx[d * 4 + h], part);
  }
}

// ---------------------------------------------------------------------------
// Pass B: ex = exp(logit - mx[dst]); denom += ex
// ---------------------------------------------------------------------------
__global__ void k_edge_expsum(const float* __restrict__ logits,
                              const int* __restrict__ dstA,
                              const float* __restrict__ mx,
                              float* __restrict__ exbuf,
                              float* __restrict__ denom) {
  int t = blockIdx.x * 256 + threadIdx.x;
  if (t >= E_TOT * 4) return;
  int e = t >> 2, h = t & 3;
  int d = dstA[e];
  float ex = __expf(logits[t] - mx[d * 4 + h]);
  exbuf[t] = ex;
  atomicAdd(&denom[d * 4 + h], ex);
}

// ---------------------------------------------------------------------------
// Pass C: agg[dst] += alpha * xl[src]  (one wave per edge, 8 ch/lane)
// ---------------------------------------------------------------------------
__global__ __launch_bounds__(256)
void k_edge_aggregate(const float* __restrict__ xl,
                      const float* __restrict__ exbuf,
                      const float* __restrict__ denom,
                      const int* __restrict__ srcA, const int* __restrict__ dstA,
                      float* __restrict__ agg) {
  const int e    = (blockIdx.x * 256 + threadIdx.x) >> 5;
  const int lane = threadIdx.x & 31;
  if (e >= E_TOT) return;
  const int s = srcA[e], d = dstA[e];

  const float4* xlp = (const float4*)(xl + (size_t)s * HC) + lane * 2;
  float4 l0 = xlp[0], l1 = xlp[1];
  float lv[8] = {l0.x, l0.y, l0.z, l0.w, l1.x, l1.y, l1.z, l1.w};

  const int cbase = lane * 8;
  const int h = cbase >> 6;                       // 8 channels stay in one head
  const float alpha = exbuf[e * 4 + h] / denom[d * 4 + h];
#pragma unroll
  for (int j = 0; j < 8; ++j) {
    atomicAdd(&agg[(size_t)d * HC + cbase + j], lv[j] * alpha);
  }
}

// ---------------------------------------------------------------------------
// Finalize GAT layer: mean over 4 heads + bias + LeakyReLU(0.01)
// ---------------------------------------------------------------------------
__global__ void k_finalize_layer(const float* __restrict__ agg,
                                 const float* __restrict__ cbias,
                                 float* __restrict__ hout) {
  int t = blockIdx.x * 256 + threadIdx.x;
  if (t >= N_NODES * 64) return;
  int n = t >> 6, c = t & 63;
  const float* a = agg + (size_t)n * HC;
  float v = 0.25f * (a[c] + a[64 + c] + a[128 + c] + a[192 + c]) + cbias[c];
  hout[t] = (v >= 0.0f) ? v : ACT_SLOPE * v;
}

// ---------------------------------------------------------------------------
// Host launch
// ---------------------------------------------------------------------------
static inline int cdiv(int a, int b) { return (a + b - 1) / b; }

static void launch_gemm(int K, const _Float16* xh, const _Float16* wt,
                        const float* bias, float* Y, int M, int Npad, int Nout,
                        int act, hipStream_t s) {
  dim3 g(M / 16, Npad / 64);
  if (K == 64)
    k_wmma_gemm<64><<<g, 128, 0, s>>>(xh, wt, bias, Y, Nout, act);
  else if (K == 128)
    k_wmma_gemm<128><<<g, 128, 0, s>>>(xh, wt, bias, Y, Nout, act);
  else
    k_wmma_gemm<256><<<g, 128, 0, s>>>(xh, wt, bias, Y, Nout, act);
}

extern "C" void kernel_launch(void* const* d_in, const int* in_sizes, int n_in,
                              void* d_out, int out_size, void* d_ws, size_t ws_size,
                              hipStream_t stream) {
  const float* x     = (const float*)d_in[0];
  const int*   ei    = (const int*)d_in[1];
  const float* eattr = (const float*)d_in[2];
  const float* Wd1   = (const float*)d_in[24];
  const float* bd1   = (const float*)d_in[25];
  const float* Wd2   = (const float*)d_in[26];
  const float* bd2   = (const float*)d_in[27];
  (void)in_sizes; (void)n_in; (void)out_size; (void)ws_size;

  // ---- workspace layout ----
  char* base = (char*)d_ws;
  auto alloc = [&](size_t bytes) -> void* {
    void* p = (void*)base;
    base += (bytes + 255) & ~(size_t)255;
    return p;
  };
  float*    ea_all = (float*)alloc((size_t)E_TOT * EDGE_DIM * 4);
  float*    xl     = (float*)alloc((size_t)N_NODES * HC * 4);
  float*    xr     = (float*)alloc((size_t)N_NODES * HC * 4);
  float*    agg    = (float*)alloc((size_t)N_NODES * HC * 4);
  float*    hbuf   = (float*)alloc((size_t)N_NODES * 64 * 4);
  float*    hd1    = (float*)alloc((size_t)N_NODES * HC * 4);
  float*    logits = (float*)alloc((size_t)E_TOT * 4 * 4);
  float*    exbuf  = (float*)alloc((size_t)E_TOT * 4 * 4);
  float*    mx     = (float*)alloc((size_t)N_NODES * 4 * 4);
  float*    denom  = (float*)alloc((size_t)N_NODES * 4 * 4);
  float*    cnt    = (float*)alloc((size_t)N_NODES * 4);
  int*      srcA   = (int*)alloc((size_t)E_TOT * 4);
  int*      dstA   = (int*)alloc((size_t)E_TOT * 4);
  _Float16* xh     = (_Float16*)alloc((size_t)N_NODES * HC * 2);  // f16 activations
  _Float16* wt     = (_Float16*)alloc((size_t)256 * 256 * 2);     // f16 W^T (padded)

  const int TB = 256;
  const float NEG_BIG = -3.402823466e38f;

  // ---- self-loop mean edge attrs + concatenated edge lists ----
  k_fill_f32<<<cdiv(N_NODES * EDGE_DIM, TB), TB, 0, stream>>>(
      ea_all + (size_t)N_EDGES * EDGE_DIM, 0.0f, N_NODES * EDGE_DIM);
  k_fill_f32<<<cdiv(N_NODES, TB), TB, 0, stream>>>(cnt, 0.0f, N_NODES);
  k_prep_edges<<<cdiv(N_EDGES * EDGE_DIM, TB), TB, 0, stream>>>(
      ei, eattr, ea_all, srcA, dstA, cnt);
  k_prep_loops<<<cdiv(N_NODES * EDGE_DIM, TB), TB, 0, stream>>>(
      ea_all, srcA, dstA, cnt);

  const int edgeWaveBlocks = cdiv(E_TOT * 32, TB);

  const float* xin = x;
  int K = 128;
  for (int layer = 0; layer < 3; ++layer) {
    const float* Wl  = (const float*)d_in[3 + layer * 7 + 0];
    const float* bl  = (const float*)d_in[3 + layer * 7 + 1];
    const float* Wr  = (const float*)d_in[3 + layer * 7 + 2];
    const float* br  = (const float*)d_in[3 + layer * 7 + 3];
    const float* We  = (const float*)d_in[3 + layer * 7 + 4];
    const float* att = (const float*)d_in[3 + layer * 7 + 5];
    const float* cb  = (const float*)d_in[3 + layer * 7 + 6];

    // f16 activations (shared by both projections)
    k_x_to_f16<<<cdiv(N_NODES * K, TB), TB, 0, stream>>>(xin, xh, N_NODES * K);

    // xl = x @ Wl + bl
    k_w_to_f16t<<<cdiv(HC * K, TB), TB, 0, stream>>>(Wl, wt, K, HC, HC);
    launch_gemm(K, xh, wt, bl, xl, N_NODES, HC, HC, 0, stream);
    // xr = x @ Wr + br
    k_w_to_f16t<<<cdiv(HC * K, TB), TB, 0, stream>>>(Wr, wt, K, HC, HC);
    launch_gemm(K, xh, wt, br, xr, N_NODES, HC, HC, 0, stream);

    // reset per-layer accumulators
    k_fill_f32<<<cdiv(N_NODES * HC, TB), TB, 0, stream>>>(agg, 0.0f, N_NODES * HC);
    k_fill_f32<<<cdiv(N_NODES * 4, TB), TB, 0, stream>>>(mx, NEG_BIG, N_NODES * 4);
    k_fill_f32<<<cdiv(N_NODES * 4, TB), TB, 0, stream>>>(denom, 0.0f, N_NODES * 4);

    // segment softmax + aggregation
    k_edge_logits<<<edgeWaveBlocks, TB, 0, stream>>>(
        xl, xr, ea_all, srcA, dstA, We, att, logits, mx);
    k_edge_expsum<<<cdiv(E_TOT * 4, TB), TB, 0, stream>>>(
        logits, dstA, mx, exbuf, denom);
    k_edge_aggregate<<<edgeWaveBlocks, TB, 0, stream>>>(
        xl, exbuf, denom, srcA, dstA, agg);

    // mean heads + bias + LeakyReLU(0.01)
    k_finalize_layer<<<cdiv(N_NODES * 64, TB), TB, 0, stream>>>(agg, cb, hbuf);

    xin = hbuf;
    K = 64;
  }

  // dense head: hd1 = act(h @ Wd1 + bd1)   [30000,64] x [64,256]
  k_x_to_f16<<<cdiv(N_NODES * 64, TB), TB, 0, stream>>>(hbuf, xh, N_NODES * 64);
  k_w_to_f16t<<<cdiv(256 * 64, TB), TB, 0, stream>>>(Wd1, wt, 64, 256, 256);
  launch_gemm(64, xh, wt, bd1, hd1, N_NODES, 256, 256, 1, stream);

  // out = hd1 @ Wd2 + bd2   [30000,256] x [256,4]  (Wt zero-padded to 64 cols)
  k_x_to_f16<<<cdiv(N_NODES * 256, TB), TB, 0, stream>>>(hd1, xh, N_NODES * 256);
  k_w_to_f16t<<<cdiv(64 * 256, TB), TB, 0, stream>>>(Wd2, wt, 256, 4, 64);
  launch_gemm(256, xh, wt, bd2, (float*)d_out, N_NODES, 64, 4, 0, stream);
}
